// MoE_91113436217559
// MI455X (gfx1250) — matrix operand order repaired and checked by reading, not compiled, not run on hardware
//
#include <hip/hip_runtime.h>
#include <hip/hip_bf16.h>

// ---------------- types & helpers ----------------
typedef __attribute__((ext_vector_type(16))) __bf16 v16bf;
typedef __attribute__((ext_vector_type(8)))  float  v8f;

union BF16Frag { v16bf v; uint4 q[2]; };

#if __has_builtin(__builtin_amdgcn_global_load_async_to_lds_b128)
#define HAVE_ASYNC_LDS 1
typedef int v4i_ __attribute__((vector_size(16)));
typedef __attribute__((address_space(1))) v4i_* gptr_v4i;
typedef __attribute__((address_space(3))) v4i_* lptr_v4i;
#else
#define HAVE_ASYNC_LDS 0
#endif

__device__ __forceinline__ unsigned short f32_to_bf16(float f) {
    unsigned u = __float_as_uint(f);
    u += 0x7FFFu + ((u >> 16) & 1u);          // round-to-nearest-even
    return (unsigned short)(u >> 16);
}

__device__ __forceinline__ float gelu_tanh(float v) {
    float c = v + 0.044715f * v * v * v;
    return 0.5f * v * (1.0f + tanhf(0.7978845608028654f * c));
}

#define NTOK 2048      // B*T
#define CDIM 1024
#define NEXP 8
#define TOPK 2
#define CAPACITY 256

// ---------------- 1) router: sigmoid(x@Wg+b), top-2, renorm gates ----------------
__global__ __launch_bounds__(256) void router_kernel(
    const float* __restrict__ x, const float* __restrict__ Wg,
    const float* __restrict__ bg, float* __restrict__ scores,
    float* __restrict__ gates, int* __restrict__ ind)
{
    int wave = threadIdx.x >> 5;
    int lane = threadIdx.x & 31;
    int t = blockIdx.x * 8 + wave;
    const float* xr = x + (size_t)t * CDIM;
    float acc[NEXP] = {0.f,0.f,0.f,0.f,0.f,0.f,0.f,0.f};
    for (int c = lane; c < CDIM; c += 32) {
        float xv = xr[c];
        const float* wr = Wg + c * NEXP;
        #pragma unroll
        for (int e = 0; e < NEXP; ++e) acc[e] += xv * wr[e];
    }
    #pragma unroll
    for (int e = 0; e < NEXP; ++e)
        #pragma unroll
        for (int off = 16; off > 0; off >>= 1)
            acc[e] += __shfl_down(acc[e], off, 32);
    if (lane == 0) {
        float s[NEXP];
        #pragma unroll
        for (int e = 0; e < NEXP; ++e) {
            float z = acc[e] + bg[e];
            s[e] = 1.0f / (1.0f + __expf(-z));
            scores[t * NEXP + e] = s[e];
        }
        float b0 = -1e30f, b1 = -1e30f; int i0 = 0, i1 = 0;
        #pragma unroll
        for (int e = 0; e < NEXP; ++e) {
            if (s[e] > b0)      { b1 = b0; i1 = i0; b0 = s[e]; i0 = e; }
            else if (s[e] > b1) { b1 = s[e]; i1 = e; }
        }
        float inv = 1.0f / (b0 + b1);
        gates[t * 2 + 0] = b0 * inv;
        gates[t * 2 + 1] = b1 * inv;
        ind[t * 2 + 0] = i0;
        ind[t * 2 + 1] = i1;
    }
}

// ---------------- 2) dispatch: priority sort + capacity slots + stats ----------------
__device__ __forceinline__ int ext16(unsigned long long lo, unsigned long long hi, int e) {
    return (int)(((e < 4) ? (lo >> (16 * e)) : (hi >> (16 * (e - 4)))) & 0xFFFFull);
}

__global__ __launch_bounds__(1024) void dispatch_kernel(
    const float* __restrict__ scores, const float* __restrict__ gates,
    const int* __restrict__ ind, int* __restrict__ slot,
    float* __restrict__ stats /* [tpe(8), f(8), p(8)] */)
{
    __shared__ unsigned long long key[NTOK];
    __shared__ unsigned char eid[NTOK * TOPK];
    __shared__ unsigned long long slo[1024], shi[1024];
    __shared__ unsigned long long base_l, base_h;
    __shared__ float p_sh[NEXP];
    int tid = threadIdx.x;

    // build sort keys: ascending u64 == (descending top-1 gate, ascending token id)
    for (int t = tid; t < NTOK; t += 1024) {
        float g0 = gates[t * 2];
        unsigned u = __float_as_uint(g0);
        unsigned m = (u & 0x80000000u) ? ~u : (u | 0x80000000u); // ascending-monotone
        unsigned d = ~m;                                         // descending-monotone
        key[t] = ((unsigned long long)d << 32) | (unsigned)t;
        eid[t * 2 + 0] = (unsigned char)ind[t * 2 + 0];
        eid[t * 2 + 1] = (unsigned char)ind[t * 2 + 1];
    }
    if (tid == 0) { base_l = 0ull; base_h = 0ull; }
    if (tid < NEXP) p_sh[tid] = 0.0f;
    __syncthreads();

    // bitonic sort (ascending u64), 2048 elements / 1024 threads
    for (unsigned k = 2; k <= NTOK; k <<= 1) {
        for (unsigned j = k >> 1; j >= 1; j >>= 1) {
            for (unsigned i = tid; i < NTOK; i += 1024) {
                unsigned ixj = i ^ j;
                if (ixj > i) {
                    bool up = ((i & k) == 0);
                    unsigned long long a = key[i], b = key[ixj];
                    if ((a > b) == up) { key[i] = b; key[ixj] = a; }
                }
            }
            __syncthreads();
        }
    }

    // slot assignment: k=0 stream over sorted order, then k=1 (matches reference flatten)
    for (int ks = 0; ks < TOPK; ++ks) {
        int s0 = tid * 2, s1 = tid * 2 + 1;
        int t0 = (int)(key[s0] & 0xFFFFFFFFull);
        int t1 = (int)(key[s1] & 0xFFFFFFFFull);
        int e0 = eid[t0 * 2 + ks], e1 = eid[t1 * 2 + ks];
        unsigned long long l0 = 0ull, h0 = 0ull;
        if (e0 < 4) l0 = 1ull << (16 * e0); else h0 = 1ull << (16 * (e0 - 4));
        unsigned long long l1 = l0, h1 = h0;
        if (e1 < 4) l1 += 1ull << (16 * e1); else h1 += 1ull << (16 * (e1 - 4));
        slo[tid] = l1; shi[tid] = h1;
        __syncthreads();
        // Hillis-Steele inclusive scan of packed 8x16-bit counters
        for (int off = 1; off < 1024; off <<= 1) {
            unsigned long long al = 0ull, ah = 0ull;
            if (tid >= off) { al = slo[tid - off]; ah = shi[tid - off]; }
            __syncthreads();
            slo[tid] += al; shi[tid] += ah;
            __syncthreads();
        }
        unsigned long long el = slo[tid] - l1, eh = shi[tid] - h1; // exclusive for pair
        int c0 = ext16(el, eh, e0) + ext16(base_l, base_h, e0);
        int c1 = ext16(el + l0, eh + h0, e1) + ext16(base_l, base_h, e1);
        slot[t0 * 2 + ks] = (c0 < CAPACITY) ? c0 : -1;
        slot[t1 * 2 + ks] = (c1 < CAPACITY) ? c1 : -1;
        __syncthreads();
        if (tid == 0) { base_l += slo[1023]; base_h += shi[1023]; }
        __syncthreads();
    }
    if (tid == 0) {
        #pragma unroll
        for (int e = 0; e < NEXP; ++e) {
            float v = (float)ext16(base_l, base_h, e) / (float)NTOK;
            stats[e] = v;        // tokens_per_expert
            stats[NEXP + e] = v; // f (identical count / BT)
        }
    }

    // p[e] = mean over tokens of scores[t,e] / rowsum(scores[t,:])
    float pa[NEXP] = {0.f,0.f,0.f,0.f,0.f,0.f,0.f,0.f};
    for (int t = tid; t < NTOK; t += 1024) {
        float sv[NEXP]; float rs = 0.f;
        #pragma unroll
        for (int e = 0; e < NEXP; ++e) { sv[e] = scores[t * NEXP + e]; rs += sv[e]; }
        float inv = 1.0f / rs;
        #pragma unroll
        for (int e = 0; e < NEXP; ++e) pa[e] += sv[e] * inv;
    }
    #pragma unroll
    for (int e = 0; e < NEXP; ++e) atomicAdd(&p_sh[e], pa[e]);
    __syncthreads();
    if (tid < NEXP) stats[2 * NEXP + tid] = p_sh[tid] / (float)NTOK;
}

// ---------------- 3) gather tokens -> bf16 expert-input buffer ----------------
__global__ __launch_bounds__(256) void gather_kernel(
    const float* __restrict__ x, const int* __restrict__ ind,
    const int* __restrict__ slot, unsigned short* __restrict__ ei)
{
    int t = blockIdx.x, tid = threadIdx.x;
    const float4* xr = (const float4*)(x + (size_t)t * CDIM);
    float4 v = xr[tid];
    union { unsigned short s[4]; uint2 u; } pk;
    pk.s[0] = f32_to_bf16(v.x); pk.s[1] = f32_to_bf16(v.y);
    pk.s[2] = f32_to_bf16(v.z); pk.s[3] = f32_to_bf16(v.w);
    #pragma unroll
    for (int k = 0; k < TOPK; ++k) {
        int s = slot[t * 2 + k];
        if (s >= 0) {
            int e = ind[t * 2 + k];
            *(uint2*)(ei + ((size_t)e * CAPACITY + s) * CDIM + tid * 4) = pk.u;
        }
    }
}

// ---------------- 4) bf16 WMMA GEMM ----------------
// Tile: 256(M) x 64(N) per 256-thread WG -> every weight element loaded ONCE from
// HBM (weights = 268 MB fp32 dominate traffic; A/h tiles live in the 192 MB L2).
// 8 waves arranged 4(M) x 2(N); each wave computes 64x32 via 4x2 WMMA accumulators.
// A (bf16) is staged with GLOBAL_LOAD_ASYNC_TO_LDS_B128 (ASYNCcnt path); B (fp32
// weights) is converted to bf16 on the fly while staging transposed into LDS.
template <bool GELU, bool OUT_BF16>
__global__ __launch_bounds__(256) void ffn_gemm(
    const unsigned short* __restrict__ A, const float* __restrict__ B,
    const float* __restrict__ bias, void* __restrict__ Out, int Kd, int N)
{
    const int e = blockIdx.z;
    const int n0 = blockIdx.x * 64;
    const int tid = threadIdx.x;
    const int lane = tid & 31;
    const int wv = tid >> 5;
    const int wm = (wv >> 1) * 64;  // 0 / 64 / 128 / 192
    const int wn = (wv & 1) * 32;   // 0 / 32

    __shared__ unsigned short lds_a[256 * 32];  // A tile, row-major [m][k]  (16 KB)
    __shared__ unsigned short lds_bt[64 * 32];  // B tile transposed [n][k]  (4 KB)

    const unsigned short* Ae = A + (size_t)e * CAPACITY * Kd;
    const float* Be = B + (size_t)e * Kd * N;

    v8f acc[4][2];
    v8f zero = {0.f,0.f,0.f,0.f,0.f,0.f,0.f,0.f};
    #pragma unroll
    for (int mi = 0; mi < 4; ++mi)
        #pragma unroll
        for (int ni = 0; ni < 2; ++ni) acc[mi][ni] = zero;

    for (int kk = 0; kk < Kd; kk += 32) {
        __syncthreads();
        // stage A: 256x32 bf16 = 1024 16-byte chunks (async DMA to LDS if available)
        #pragma unroll
        for (int i = 0; i < 4; ++i) {
            int c = tid + i * 256;
            int row = c >> 2, kc = (c & 3) * 8;
            const unsigned short* src = Ae + (size_t)row * Kd + kk + kc;
#if HAVE_ASYNC_LDS
            __builtin_amdgcn_global_load_async_to_lds_b128(
                (gptr_v4i)src, (lptr_v4i)&lds_a[row * 32 + kc], 0, 0);
#else
            *(uint4*)&lds_a[row * 32 + kc] = *(const uint4*)src;
#endif
        }
        // stage B transposed: 32x64 fp32 -> bf16 [n][k]; prefetch next k-step rows
        #pragma unroll
        for (int i = 0; i < 2; ++i) {
            int c = tid + i * 256;
            int row = c >> 4, nc = (c & 15) * 4;
            const float* src = Be + (size_t)(kk + row) * N + n0 + nc;
            float4 f = *(const float4*)src;
            if (kk + 32 < Kd) __builtin_prefetch(src + (size_t)32 * N, 0, 1);
            lds_bt[(nc + 0) * 32 + row] = f32_to_bf16(f.x);
            lds_bt[(nc + 1) * 32 + row] = f32_to_bf16(f.y);
            lds_bt[(nc + 2) * 32 + row] = f32_to_bf16(f.z);
            lds_bt[(nc + 3) * 32 + row] = f32_to_bf16(f.w);
        }
#if HAVE_ASYNC_LDS
        asm volatile("s_wait_asynccnt 0x0" ::: "memory");
#endif
        __syncthreads();

        // fragments per ISA 16-bit A(16x32)/B(32x16) wave32 layouts
        BF16Frag afr[4], bfr[2];
        int rowb = lane & 15;
        int kbA = (lane >> 4) * 8;   // A: lanes 0-15 K{0..7,16..23}; 16-31 K{8..15,24..31}
        #pragma unroll
        for (int mi = 0; mi < 4; ++mi) {
            int r = wm + mi * 16 + rowb;
            afr[mi].q[0] = *(const uint4*)(&lds_a[r * 32 + kbA]);
            afr[mi].q[1] = *(const uint4*)(&lds_a[r * 32 + kbA + 16]);
        }
        int kbB = (lane >> 4) * 8;   // B: lanes 0-15 K{0..7,16..23}; 16-31 K{8..15,24..31} (col = lane&15)
        #pragma unroll
        for (int ni = 0; ni < 2; ++ni) {
            int cc = wn + ni * 16 + rowb;
            bfr[ni].q[0] = *(const uint4*)(&lds_bt[cc * 32 + kbB]);
            bfr[ni].q[1] = *(const uint4*)(&lds_bt[cc * 32 + kbB + 16]);
        }
        #pragma unroll
        for (int mi = 0; mi < 4; ++mi)
            #pragma unroll
            for (int ni = 0; ni < 2; ++ni)
                acc[mi][ni] = __builtin_amdgcn_wmma_f32_16x16x32_bf16(
                    false, afr[mi].v, false, bfr[ni].v,
                    (short)0, acc[mi][ni], false, false);
    }

    // epilogue: +bias, optional gelu, store (C/D layout: row = v + (lane>>4)*8, col = lane&15)
    const float* be = bias + (size_t)e * N;
    int colL = lane & 15;
    int rAdd = (lane >> 4) * 8;
    #pragma unroll
    for (int mi = 0; mi < 4; ++mi) {
        #pragma unroll
        for (int ni = 0; ni < 2; ++ni) {
            int gcol = n0 + wn + ni * 16 + colL;
            float bv = be[gcol];
            #pragma unroll
            for (int v = 0; v < 8; ++v) {
                int grow = wm + mi * 16 + v + rAdd;
                float val = acc[mi][ni][v] + bv;
                if (GELU) val = gelu_tanh(val);
                size_t o = ((size_t)e * CAPACITY + grow) * N + gcol;
                if (OUT_BF16) ((unsigned short*)Out)[o] = f32_to_bf16(val);
                else          ((float*)Out)[o] = val;
            }
        }
    }
}

// ---------------- 5) combine: out[t] = sum_k gate * eo[e_k, slot_k] ----------------
__global__ __launch_bounds__(256) void combine_kernel(
    const float* __restrict__ eo, const float* __restrict__ gates,
    const int* __restrict__ ind, const int* __restrict__ slot,
    float* __restrict__ out)
{
    int t = blockIdx.x, tid = threadIdx.x;
    float4 a = {0.f, 0.f, 0.f, 0.f};
    #pragma unroll
    for (int k = 0; k < TOPK; ++k) {
        int s = slot[t * 2 + k];
        if (s >= 0) {
            int e = ind[t * 2 + k];
            float g = gates[t * 2 + k];
            float4 v = ((const float4*)(eo + ((size_t)e * CAPACITY + s) * CDIM))[tid];
            a.x += g * v.x; a.y += g * v.y; a.z += g * v.z; a.w += g * v.w;
        }
    }
    ((float4*)(out + (size_t)t * CDIM))[tid] = a;
}

// ---------------- launch ----------------
extern "C" void kernel_launch(void* const* d_in, const int* in_sizes, int n_in,
                              void* d_out, int out_size, void* d_ws, size_t ws_size,
                              hipStream_t stream) {
    (void)in_sizes; (void)n_in; (void)out_size; (void)ws_size;
    const float* x  = (const float*)d_in[0];
    // d_in[1]/d_in[2] (W_shared/b_shared): computed-then-discarded in reference; skipped.
    const float* Wg = (const float*)d_in[3];
    const float* bg = (const float*)d_in[4];
    const float* W1 = (const float*)d_in[5];
    const float* b1 = (const float*)d_in[6];
    const float* W2 = (const float*)d_in[7];
    const float* b2 = (const float*)d_in[8];
    float* out = (float*)d_out;

    char* ws = (char*)d_ws;
    float*          scores = (float*)(ws + 0);                 // 64 KB
    float*          gates  = (float*)(ws + (1 << 16));         // 16 KB
    int*            ind    = (int*)  (ws + 81920);             // 16 KB
    int*            slot   = (int*)  (ws + 98304);             // 16 KB
    unsigned short* ei     = (unsigned short*)(ws + 131072);               // 4 MB bf16
    unsigned short* h      = (unsigned short*)(ws + 131072 + 4194304);     // 16 MB bf16
    float*          eo     = (float*)(ws + 131072 + 4194304 + 16777216);   // 8 MB f32

    router_kernel<<<NTOK / 8, 256, 0, stream>>>(x, Wg, bg, scores, gates, ind);
    dispatch_kernel<<<1, 1024, 0, stream>>>(scores, gates, ind, slot,
                                            out + (size_t)NTOK * CDIM);
    (void)hipMemsetAsync(ei, 0, (size_t)NEXP * CAPACITY * CDIM * 2, stream);
    gather_kernel<<<NTOK, 256, 0, stream>>>(x, ind, slot, ei);
    ffn_gemm<true,  true ><<<dim3(64, 1, NEXP), 256, 0, stream>>>(ei, W1, b1, (void*)h, 1024, 4096);
    ffn_gemm<false, false><<<dim3(16, 1, NEXP), 256, 0, stream>>>(h,  W2, b2, (void*)eo, 4096, 1024);
    combine_kernel<<<NTOK, 256, 0, stream>>>(eo, gates, ind, slot, out);
}
